// SimclrLoss_1941325218122
// MI455X (gfx1250) — matrix operand order repaired and checked by reading, not compiled, not run on hardware
//
#include <hip/hip_runtime.h>
#include <hip/hip_bf16.h>

typedef __attribute__((ext_vector_type(16))) __bf16 v16bf;
typedef __attribute__((ext_vector_type(8)))  float  v8f;

#define B_ROWS 4096
#define DDIM   128
#define NROWS  8192
#define NTILES 512   // 8192 / 16 column tiles

// sqrt( (1/TEMP) * log2(e) ) = sqrt(2 * 1.4426950408889634)
#define ZSCALE 1.69864360057604f
#define LN2F   0.69314718055994531f

union TileU { v16bf v; uint4 q[2]; };

__device__ __forceinline__ unsigned short f2bf(float f) {
  unsigned u = __builtin_bit_cast(unsigned, f);
  u += 0x7FFFu + ((u >> 16) & 1u);          // round-to-nearest-even
  return (unsigned short)(u >> 16);
}

__device__ __forceinline__ float fexp2(float x) {
#if __has_builtin(__builtin_amdgcn_exp2f)
  return __builtin_amdgcn_exp2f(x);
#else
  return exp2f(x);
#endif
}

// Kernel 1: L2-normalize rows of zi/zj, fold sqrt(log2e/TEMP), pack to bf16.
// One wave per row: 32 lanes x float4 = 128 floats.
__global__ void __launch_bounds__(256)
simclr_normalize(const float* __restrict__ zi, const float* __restrict__ zj,
                 unsigned short* __restrict__ Z) {
  const int wave = threadIdx.x >> 5;
  const int lane = threadIdx.x & 31;
  const int row  = blockIdx.x * 8 + wave;            // 0..8191
  const float* src = (row < B_ROWS) ? (zi + (size_t)row * DDIM)
                                    : (zj + (size_t)(row - B_ROWS) * DDIM);
  float4 x = ((const float4*)src)[lane];
  float ss = x.x * x.x + x.y * x.y + x.z * x.z + x.w * x.w;
#pragma unroll
  for (int m = 1; m < 32; m <<= 1) ss += __shfl_xor(ss, m, 32);
  float n = sqrtf(ss);
  float scale = ZSCALE / fmaxf(n, 1e-12f);
  unsigned b0 = f2bf(x.x * scale), b1 = f2bf(x.y * scale);
  unsigned b2 = f2bf(x.z * scale), b3 = f2bf(x.w * scale);
  uint2 pk; pk.x = b0 | (b1 << 16); pk.y = b2 | (b3 << 16);
  *(uint2*)(Z + (size_t)row * DDIM + lane * 4) = pk;
}

// Load one 16-col x K=128 B tile as four K=32 chunks (rows c0..c0+15 of Z, Gram symmetry).
__device__ __forceinline__ void loadB(const unsigned short* __restrict__ Z,
                                      int c0, int lane, bool hi, TileU b[4]) {
  const unsigned short* zc = Z + (size_t)(c0 + (lane & 15)) * DDIM;
#pragma unroll
  for (int kc = 0; kc < 4; ++kc) {
    const int kb = kc * 32 + (hi ? 8 : 0);
    b[kc].q[0] = *(const uint4*)(zc + kb);
    b[kc].q[1] = *(const uint4*)(zc + kb + 16);
  }
}

// Rare-path fixup: exclude self-similarity diagonal / capture positive logit.
__device__ __forceinline__ void fixup(int t, int rb, int posTile, int lane, bool hi,
                                      const v8f& acc, const float ev[8],
                                      float rs[8], float& myPos) {
#pragma unroll
  for (int v = 0; v < 8; ++v) {
    const int m = v + (hi ? 8 : 0);                  // row of elem v; col N = lane&15
    if ((lane & 15) == m) {                          // tile-diagonal element
      if (t == rb)      rs[v] -= ev[v];              // undo self-similarity term
      if (t == posTile) myPos = acc[v];              // log2e-scaled positive
    }
  }
}

// Kernel 2: fused bf16-WMMA Gram row-block + exp2 row-sum + positive capture.
__global__ void __launch_bounds__(256)
simclr_rows(const unsigned short* __restrict__ Z, float* __restrict__ partial) {
  __shared__ float sumLds[16][8];
  __shared__ float posLds[16];
  const int lane = threadIdx.x & 31;
  const int wv   = __builtin_amdgcn_readfirstlane(threadIdx.x >> 5);  // scalar wave id
  const int rb   = blockIdx.x;                       // 0..511
  const int r0   = rb * 16;
  const bool hi  = lane >= 16;

  // A tiles (16 rows x K=128) in WMMA bf16 A layout:
  // lanes 0-15: M=lane, elems = K{0..7,16..23}; lanes 16-31: K{8..15,24..31}
  TileU a[4];
  {
    const unsigned short* zr = Z + (size_t)(r0 + (lane & 15)) * DDIM;
#pragma unroll
    for (int kc = 0; kc < 4; ++kc) {
      const int kb = kc * 32;
      a[kc].q[0] = *(const uint4*)(zr + kb + (hi ? 8 : 0));
      a[kc].q[1] = *(const uint4*)(zr + kb + (hi ? 24 : 16));
    }
  }

  float rs[8] = {0.f, 0.f, 0.f, 0.f, 0.f, 0.f, 0.f, 0.f};
  float myPos = 0.f;
  const int posTile = rb ^ (B_ROWS / 16);            // rb ^ 256; 256%8==0 -> same wave as rb

  // Unroll-by-2 over this wave's 64 column tiles: two independent WMMA chains
  // per iteration hide the WMMA->VALU hazard behind the partner tile's work.
  for (int k = 0; k < 64; k += 2) {
    const int t0 = wv + (k << 3);
    const int t1 = t0 + 8;
    TileU b0[4], b1[4];
    loadB(Z, t0 * 16, lane, hi, b0);
    loadB(Z, t1 * 16, lane, hi, b1);

    v8f acc0 = {0.f, 0.f, 0.f, 0.f, 0.f, 0.f, 0.f, 0.f};
    v8f acc1 = {0.f, 0.f, 0.f, 0.f, 0.f, 0.f, 0.f, 0.f};
#pragma unroll
    for (int kc = 0; kc < 4; ++kc)
      acc0 = __builtin_amdgcn_wmma_f32_16x16x32_bf16(false, a[kc].v, false, b0[kc].v,
                                                     (short)0, acc0, false, false);
#pragma unroll
    for (int kc = 0; kc < 4; ++kc)
      acc1 = __builtin_amdgcn_wmma_f32_16x16x32_bf16(false, a[kc].v, false, b1[kc].v,
                                                     (short)0, acc1, false, false);

    float ev0[8], ev1[8];
#pragma unroll
    for (int v = 0; v < 8; ++v) { ev0[v] = fexp2(acc0[v]); rs[v] += ev0[v]; }
#pragma unroll
    for (int v = 0; v < 8; ++v) { ev1[v] = fexp2(acc1[v]); rs[v] += ev1[v]; }

    // Wave-uniform rare branches (2 of 64 tiles per wave at most)
    if (t0 == rb || t0 == posTile) fixup(t0, rb, posTile, lane, hi, acc0, ev0, rs, myPos);
    if (t1 == rb || t1 == posTile) fixup(t1, rb, posTile, lane, hi, acc1, ev1, rs, myPos);
  }

  // Sum each row across the 16 lanes (columns) of its half
#pragma unroll
  for (int v = 0; v < 8; ++v) {
    float s = rs[v];
    s += __shfl_xor(s, 1, 32);
    s += __shfl_xor(s, 2, 32);
    s += __shfl_xor(s, 4, 32);
    s += __shfl_xor(s, 8, 32);
    rs[v] = s;
  }
  if ((lane & 15) == 0) {
#pragma unroll
    for (int v = 0; v < 8; ++v) sumLds[v + (hi ? 8 : 0)][wv] = rs[v];
  }
  if (wv == (rb & 7)) {                              // wave that saw the positive tile
    if (lane < 8 || lane >= 24) posLds[lane & 15] = myPos;
  }
  __syncthreads();

  float l = 0.f;
  if (threadIdx.x < 16) {
    float tot = 0.f;
#pragma unroll
    for (int w = 0; w < 8; ++w) tot += sumLds[threadIdx.x][w];
    l = __logf(tot) - posLds[threadIdx.x] * LN2F;    // un-scale log2e positive
  }
  if (threadIdx.x < 32) {
    l += __shfl_xor(l, 1, 32);
    l += __shfl_xor(l, 2, 32);
    l += __shfl_xor(l, 4, 32);
    l += __shfl_xor(l, 8, 32);
    if (threadIdx.x == 0) partial[rb] = l;
  }
}

// Kernel 3: reduce 512 block partials -> mean loss
__global__ void __launch_bounds__(512)
simclr_finish(const float* __restrict__ partial, float* __restrict__ out) {
  __shared__ float sm[512];
  const int t = threadIdx.x;
  sm[t] = partial[t];
  __syncthreads();
  for (int s = 256; s > 0; s >>= 1) {
    if (t < s) sm[t] += sm[t + s];
    __syncthreads();
  }
  if (t == 0) out[0] = sm[0] * (1.0f / (float)NROWS);
}

extern "C" void kernel_launch(void* const* d_in, const int* in_sizes, int n_in,
                              void* d_out, int out_size, void* d_ws, size_t ws_size,
                              hipStream_t stream) {
  (void)in_sizes; (void)n_in; (void)out_size; (void)ws_size;
  const float* zi = (const float*)d_in[0];
  const float* zj = (const float*)d_in[1];
  unsigned short* Z = (unsigned short*)d_ws;                        // 8192x128 bf16 = 2 MB
  float* partial = (float*)((char*)d_ws + (size_t)NROWS * DDIM * sizeof(unsigned short));
  simclr_normalize<<<NROWS / 8, 256, 0, stream>>>(zi, zj, Z);
  simclr_rows<<<NTILES, 256, 0, stream>>>(Z, partial);
  simclr_finish<<<1, 512, 0, stream>>>(partial, (float*)d_out);
}
